// PointCloud3DCNN_52647709114951
// MI455X (gfx1250) — matrix operand order repaired and checked by reading, not compiled, not run on hardware
//
#include <hip/hip_runtime.h>

typedef __attribute__((ext_vector_type(16))) _Float16 v16h;
typedef __attribute__((ext_vector_type(8)))  _Float16 v8h;
typedef __attribute__((ext_vector_type(8)))  float    v8f;

#define EPSBN 1e-5f

// ---------------------------------------------------------------------------
// Implicit-GEMM 3D conv via V_WMMA_F32_16X16X32_F16.
// One wave = 32 output voxels (2 M-tiles) x 16 output channels; each B
// fragment feeds two independent WMMAs. Gather validity is fully branch-free:
// nested compile-time k-loops + unsigned range checks + zero-page clamp.
// MODE: 0 = stride1 pad1, 1 = stride2 pad1, 2 = transposed (k3,s2,p1 inverse)
// Weights pre-packed as wt[co][tap*CINP + ci] (f16, BN scale folded).
// Epilogue: +bias (folded BN), ReLU, * mask, store f16.
// ---------------------------------------------------------------------------
struct ConvArgs {
  const _Float16* __restrict__ in;
  _Float16* __restrict__ out;
  const _Float16* __restrict__ wt;
  const float* __restrict__ bias;
  const _Float16* __restrict__ mask;   // at output resolution
  const _Float16* __restrict__ zpage;  // >= 2*CINP bytes of zeros
  int B, Di, Hi, Wi, CinP;
  int Do, Ho, Wo, CoutP, NP;
  int mode;
};

__device__ __forceinline__ void decomp(int v, int Wo, int Ho, int Do,
                                       int& b, int& d, int& h, int& w) {
  w = v % Wo; int t = v / Wo;
  h = t % Ho; t /= Ho;
  d = t % Do; b = t / Do;
}

template <int CINP, int MODE>
__global__ __launch_bounds__(32) void conv3d_wmma(ConvArgs a) {
  const int lane = threadIdx.x;
  const int row  = lane & 15;   // A-row within tile; also B/C column
  const int hi   = lane >> 4;   // K-half selector
  const int Mtot = a.B * a.Do * a.Ho * a.Wo;
  const int m0   = blockIdx.x * 32;
  const int colch = blockIdx.y * 16 + row;

  // two M-tiles per wave
  const int v0 = m0 + row, v1 = m0 + 16 + row;
  const int rv0 = (v0 < Mtot), rv1 = (v1 < Mtot);
  int b0_, d0_, h0_, w0_, b1_, d1_, h1_, w1_;
  decomp(rv0 ? v0 : 0, a.Wo, a.Ho, a.Do, b0_, d0_, h0_, w0_);
  decomp(rv1 ? v1 : 0, a.Wo, a.Ho, a.Do, b1_, d1_, h1_, w1_);

  const long sH = (long)a.Wi * CINP;          // halves per H-line
  const long sD = (long)a.Hi * sH;            // halves per D-slice
  const _Float16* __restrict__ base0 = a.in + (long)b0_ * a.Di * sD + hi * 8;
  const _Float16* __restrict__ base1 = a.in + (long)b1_ * a.Di * sD + hi * 8;
  const _Float16* __restrict__ zp = a.zpage;
  const _Float16* __restrict__ wrow = a.wt + (long)colch * (27L * CINP) + hi * 8;

  const unsigned uDi = (unsigned)a.Di, uHi = (unsigned)a.Hi, uWi = (unsigned)a.Wi;
  const int S = (MODE == 1) ? 2 : 1;

  v8f acc0, acc1;
  {
    const float bv = a.bias[colch];
#pragma unroll
    for (int j = 0; j < 8; ++j) { acc0[j] = bv; acc1[j] = bv; }
  }

  for (int kd = 0; kd < 3; ++kd) {
    int id0, id1, vd0, vd1;
    if (MODE == 2) {
      const int td0 = d0_ - 1 + kd, td1 = d1_ - 1 + kd;
      id0 = td0 >> 1; id1 = td1 >> 1;
      vd0 = rv0 & ((td0 & 1) ^ 1) & (int)((unsigned)id0 < uDi);
      vd1 = rv1 & ((td1 & 1) ^ 1) & (int)((unsigned)id1 < uDi);
    } else {
      id0 = d0_ * S - 1 + kd; id1 = d1_ * S - 1 + kd;
      vd0 = rv0 & (int)((unsigned)id0 < uDi);
      vd1 = rv1 & (int)((unsigned)id1 < uDi);
    }
    const _Float16* __restrict__ pd0 = base0 + (long)id0 * sD;
    const _Float16* __restrict__ pd1 = base1 + (long)id1 * sD;

    for (int kh = 0; kh < 3; ++kh) {
      int ih0, ih1, vh0, vh1;
      if (MODE == 2) {
        const int th0 = h0_ - 1 + kh, th1 = h1_ - 1 + kh;
        ih0 = th0 >> 1; ih1 = th1 >> 1;
        vh0 = vd0 & ((th0 & 1) ^ 1) & (int)((unsigned)ih0 < uHi);
        vh1 = vd1 & ((th1 & 1) ^ 1) & (int)((unsigned)ih1 < uHi);
      } else {
        ih0 = h0_ * S - 1 + kh; ih1 = h1_ * S - 1 + kh;
        vh0 = vd0 & (int)((unsigned)ih0 < uHi);
        vh1 = vd1 & (int)((unsigned)ih1 < uHi);
      }
      const _Float16* __restrict__ pdh0 = pd0 + (long)ih0 * sH;
      const _Float16* __restrict__ pdh1 = pd1 + (long)ih1 * sH;

      for (int kw = 0; kw < 3; ++kw) {
        int iw0, iw1, t0, t1;
        if (MODE == 2) {
          const int tw0 = w0_ - 1 + kw, tw1 = w1_ - 1 + kw;
          iw0 = tw0 >> 1; iw1 = tw1 >> 1;
          t0 = vh0 & ((tw0 & 1) ^ 1) & (int)((unsigned)iw0 < uWi);
          t1 = vh1 & ((tw1 & 1) ^ 1) & (int)((unsigned)iw1 < uWi);
        } else {
          iw0 = w0_ * S - 1 + kw; iw1 = w1_ * S - 1 + kw;
          t0 = vh0 & (int)((unsigned)iw0 < uWi);
          t1 = vh1 & (int)((unsigned)iw1 < uWi);
        }
        // branch-free: clamp invalid lanes onto the zero page
        const _Float16* __restrict__ pa0 = t0 ? pdh0 + (long)iw0 * CINP : zp;
        const _Float16* __restrict__ pa1 = t1 ? pdh1 + (long)iw1 * CINP : zp;
        const _Float16* __restrict__ pw = wrow + (long)(kd * 9 + kh * 3 + kw) * CINP;
        __builtin_prefetch(pw + CINP, 0, 0);  // next tap's weight line

#pragma unroll
        for (int c0 = 0; c0 < CINP; c0 += 32) {
          // A: lane<16 -> row, K{0..7,16..23}; lane>=16 -> K{8..15,24..31}
          const v8h a00 = *(const v8h*)(pa0 + c0);
          const v8h a01 = *(const v8h*)(pa0 + c0 + 16);
          const v8h a10 = *(const v8h*)(pa1 + c0);
          const v8h a11 = *(const v8h*)(pa1 + c0 + 16);
          // B: lane<16 -> col, K{0..7,16..23}; lane>=16 -> K{8..15,24..31}
          const v8h bb0 = *(const v8h*)(pw + c0);
          const v8h bb1 = *(const v8h*)(pw + c0 + 16);
          const v16h A0 = __builtin_shufflevector(a00, a01, 0, 1, 2, 3, 4, 5, 6, 7,
                                                  8, 9, 10, 11, 12, 13, 14, 15);
          const v16h A1 = __builtin_shufflevector(a10, a11, 0, 1, 2, 3, 4, 5, 6, 7,
                                                  8, 9, 10, 11, 12, 13, 14, 15);
          const v16h Bf = __builtin_shufflevector(bb0, bb1, 0, 1, 2, 3, 4, 5, 6, 7,
                                                  8, 9, 10, 11, 12, 13, 14, 15);
          acc0 = __builtin_amdgcn_wmma_f32_16x16x32_f16(false, A0, false, Bf,
                                                        (short)0, acc0, false, false);
          acc1 = __builtin_amdgcn_wmma_f32_16x16x32_f16(false, A1, false, Bf,
                                                        (short)0, acc1, false, false);
        }
      }
    }
  }

  // D layout: lane<16 = col lane rows 0..7; lane>=16 = col lane-16 rows 8..15
  const int mbase = hi * 8;
#pragma unroll
  for (int j = 0; j < 8; ++j) {
    const int vox = m0 + mbase + j;
    if (vox < Mtot) {
      float val = acc0[j];
      val = val > 0.0f ? val : 0.0f;
      val *= (float)a.mask[vox];
      a.out[(long)vox * a.CoutP + colch] = (_Float16)val;
    }
  }
#pragma unroll
  for (int j = 0; j < 8; ++j) {
    const int vox = m0 + 16 + mbase + j;
    if (vox < Mtot) {
      float val = acc1[j];
      val = val > 0.0f ? val : 0.0f;
      val *= (float)a.mask[vox];
      a.out[(long)vox * a.CoutP + colch] = (_Float16)val;
    }
  }
}

// --- input conversion f32->f16 (pad 9 -> 32 ch via prior memset) + mask ----
__global__ void prep_input_k(const float* __restrict__ x, _Float16* __restrict__ xin,
                             _Float16* __restrict__ m0, int nvox) {
  const int v = blockIdx.x * blockDim.x + threadIdx.x;
  if (v >= nvox) return;
  const float* xp = x + (long)v * 9;
  float any = 0.0f;
#pragma unroll
  for (int c = 0; c < 9; ++c) {
    const float f = xp[c];
    any += fabsf(f);
    xin[(long)v * 32 + c] = (_Float16)f;
  }
  m0[v] = (_Float16)(any != 0.0f ? 1.0f : 0.0f);
}

// --- fold BN into weights: wt[co][tap*CinP+ci] = w * gamma*rsqrt(var+eps) ---
__global__ void prep_weights_k(const float* __restrict__ wsrc, const float* __restrict__ gamma,
                               const float* __restrict__ beta, const float* __restrict__ mean,
                               const float* __restrict__ var,
                               _Float16* __restrict__ wt, float* __restrict__ bias,
                               int Cin, int Cout, int CinP, int NP) {
  const long idx = blockIdx.x * (long)blockDim.x + threadIdx.x;
  const long total = (long)NP * 27 * CinP;
  if (idx < total) {
    const int ci = (int)(idx % CinP);
    const int tp = (int)((idx / CinP) % 27);
    const int co = (int)(idx / ((long)CinP * 27));
    float val = 0.0f;
    if (ci < Cin && co < Cout) {
      const float s = gamma[co] * rsqrtf(var[co] + EPSBN);
      val = wsrc[((long)tp * Cin + ci) * Cout + co] * s;
    }
    wt[idx] = (_Float16)val;  // idx == co*27*CinP + tp*CinP + ci
  }
  if (idx < NP) {
    float bv = 0.0f;
    if (idx < Cout) {
      const float s = gamma[idx] * rsqrtf(var[idx] + EPSBN);
      bv = beta[idx] - mean[idx] * s;
    }
    bias[idx] = bv;
  }
}

// --- 3x3x3 max-pool stride2 pad1 on the mask --------------------------------
__global__ void mask_down_k(const _Float16* __restrict__ mi, _Float16* __restrict__ mo,
                            int B, int Di, int Hi, int Wi, int Do, int Ho, int Wo) {
  const long n = (long)B * Do * Ho * Wo;
  const long v = blockIdx.x * (long)blockDim.x + threadIdx.x;
  if (v >= n) return;
  int w = (int)(v % Wo); long t = v / Wo;
  int h = (int)(t % Ho); t /= Ho;
  int d = (int)(t % Do); int b = (int)(t / Do);
  float mx = 0.0f;
  for (int kd = 0; kd < 3; ++kd) { int id = 2*d - 1 + kd; if (id < 0 || id >= Di) continue;
    for (int kh = 0; kh < 3; ++kh) { int ih = 2*h - 1 + kh; if (ih < 0 || ih >= Hi) continue;
      for (int kw = 0; kw < 3; ++kw) { int iw = 2*w - 1 + kw; if (iw < 0 || iw >= Wi) continue;
        mx = fmaxf(mx, (float)mi[(((long)b * Di + id) * Hi + ih) * Wi + iw]);
      } } }
  mo[v] = (_Float16)(mx > 0.0f ? 1.0f : 0.0f);
}

// --- elementwise skip add ---------------------------------------------------
__global__ void add_k(_Float16* __restrict__ dst, const _Float16* __restrict__ src, long n) {
  const long i = blockIdx.x * (long)blockDim.x + threadIdx.x;
  if (i < n) dst[i] = (_Float16)((float)dst[i] + (float)src[i]);
}

// --- cls head: pointwise linear -> softmax[1] * mask ------------------------
__global__ void cls_k(const _Float16* __restrict__ d, const _Float16* __restrict__ mask,
                      const float* __restrict__ Wc, const float* __restrict__ bc,
                      float* __restrict__ out, int nvox, int C, int CP) {
  const int v = blockIdx.x * blockDim.x + threadIdx.x;
  if (v >= nvox) return;
  float l0 = bc[0], l1 = bc[1], l2 = bc[2];
  const _Float16* dp = d + (long)v * CP;
  for (int c = 0; c < C; ++c) {
    const float f = (float)dp[c];
    l0 += f * Wc[c * 3 + 0]; l1 += f * Wc[c * 3 + 1]; l2 += f * Wc[c * 3 + 2];
  }
  const float mx = fmaxf(l0, fmaxf(l1, l2));
  const float e0 = expf(l0 - mx), e1 = expf(l1 - mx), e2 = expf(l2 - mx);
  out[v] = (e1 / (e0 + e1 + e2)) * (float)mask[v];
}

// --- occupancy head: dense conv 16->1 + sigmoid -----------------------------
__global__ void occ_k(const _Float16* __restrict__ d0, const float* __restrict__ w,
                      const float* __restrict__ bptr, float* __restrict__ out,
                      int B, int D, int H, int W) {
  const int n = B * D * H * W;
  const int v = blockIdx.x * blockDim.x + threadIdx.x;
  if (v >= n) return;
  int ww = v % W; int t = v / W;
  int hh = t % H; t /= H;
  int dd = t % D; int bb = t / D;
  float s = bptr[0];
  for (int kd = 0; kd < 3; ++kd) { int id = dd - 1 + kd; if (id < 0 || id >= D) continue;
    for (int kh = 0; kh < 3; ++kh) { int ih = hh - 1 + kh; if (ih < 0 || ih >= H) continue;
      for (int kw = 0; kw < 3; ++kw) { int iw = ww - 1 + kw; if (iw < 0 || iw >= W) continue;
        const _Float16* ip = d0 + ((((long)bb * D + id) * H + ih) * W + iw) * 32;
        const float* wp = w + ((kd * 3 + kh) * 3 + kw) * 16;
#pragma unroll
        for (int c = 0; c < 16; ++c) s += (float)ip[c] * wp[c];
      } } }
  out[v] = 1.0f / (1.0f + expf(-s));
}

// --- coordinate regression head --------------------------------------------
__global__ void preds_k(const _Float16* __restrict__ feat, float* __restrict__ out,
                        int B, int D, int H, int W) {
  const int n = B * D * H * W;
  const int v = blockIdx.x * blockDim.x + threadIdx.x;
  if (v >= n) return;
  int ww = v % W; int t = v / W;
  int hh = t % H; t /= H;
  int dd = t % D; int bb = t / D;
  float ctr[3];
  ctr[0] = ww * 0.05f - 3.0f + 0.025f;
  ctr[1] = hh * 0.05f - 3.0f + 0.025f;
  ctr[2] = dd * 0.05f - 1.0f + 0.025f;
  const _Float16* fp = feat + (long)v * 32;
  const long base = (long)bb * D * H * W * 9 + (long)((dd * H + hh) * W + ww) * 9;
#pragma unroll
  for (int p = 0; p < 3; ++p)
#pragma unroll
    for (int cc = 0; cc < 3; ++cc) {
      const float f = (float)fp[p * 3 + cc];
      out[base + p * 3 + cc] = (f == 0.0f) ? 0.0f : (ctr[cc] + f) * 20.0f;  // /0.05
    }
}

// --- dispatch over compile-time (CINP, MODE) --------------------------------
static void launch_conv(const ConvArgs& a, hipStream_t stream) {
  const int Mtot = a.B * a.Do * a.Ho * a.Wo;
  dim3 grid((unsigned)((Mtot + 31) / 32), (unsigned)(a.NP / 16));
  dim3 blk(32);
#define CONV_CASE(C, M)                                                         \
  if (a.CinP == C && a.mode == M) {                                             \
    hipLaunchKernelGGL((conv3d_wmma<C, M>), grid, blk, 0, stream, a);           \
    return;                                                                     \
  }
  CONV_CASE(32, 0)  CONV_CASE(32, 1)  CONV_CASE(32, 2)
  CONV_CASE(64, 0)  CONV_CASE(64, 1)  CONV_CASE(64, 2)
  CONV_CASE(128, 0) CONV_CASE(128, 1) CONV_CASE(128, 2)
  CONV_CASE(256, 0) CONV_CASE(256, 1) CONV_CASE(256, 2)
#undef CONV_CASE
}

// ===========================================================================
extern "C" void kernel_launch(void* const* d_in, const int* in_sizes, int n_in,
                              void* d_out, int out_size, void* d_ws, size_t ws_size,
                              hipStream_t stream) {
  (void)in_sizes; (void)n_in; (void)out_size; (void)ws_size;

  const int B = 2;
  const int D0 = 64, H0 = 96, W0 = 96;
  const int D1 = 32, H1 = 48, W1 = 48;
  const int D2 = 16, H2 = 24, W2 = 24;
  const int D3 = 8,  H3 = 12, W3 = 12;
  const int D4 = 4,  H4 = 6,  W4 = 6;
  const long nv0 = (long)B * D0 * H0 * W0;
  const long nv1 = (long)B * D1 * H1 * W1;
  const long nv2 = (long)B * D2 * H2 * W2;
  const long nv3 = (long)B * D3 * H3 * W3;
  const long nv4 = (long)B * D4 * H4 * W4;

  // ---------------- workspace layout (bump allocator, 256B aligned) --------
  char* p = (char*)d_ws;
  auto alloc = [&](size_t bytes) -> char* {
    char* r = p; p += (bytes + 255) & ~(size_t)255; return r;
  };
  _Float16* zpage = (_Float16*)alloc(1024);                 // stays all-zero
  _Float16* bufA = (_Float16*)alloc((size_t)nv0 * 32 * 2);  // xin -> dec2_inv -> dec1 feat
  _Float16* bufB = (_Float16*)alloc((size_t)nv0 * 32 * 2);  // enc1 out -> dec0
  _Float16* tmp1 = (_Float16*)alloc((size_t)nv1 * 32 * 2);
  _Float16* fe2  = (_Float16*)alloc((size_t)nv1 * 32 * 2);
  _Float16* dq1  = (_Float16*)alloc((size_t)nv1 * 32 * 2);
  _Float16* tmp2 = (_Float16*)alloc((size_t)nv2 * 64 * 2);
  _Float16* fe3  = (_Float16*)alloc((size_t)nv2 * 64 * 2);
  _Float16* dq2  = (_Float16*)alloc((size_t)nv2 * 64 * 2);
  _Float16* tmp3 = (_Float16*)alloc((size_t)nv3 * 128 * 2);
  _Float16* fe4  = (_Float16*)alloc((size_t)nv3 * 128 * 2);
  _Float16* dq3  = (_Float16*)alloc((size_t)nv3 * 128 * 2);
  _Float16* tmp4 = (_Float16*)alloc((size_t)nv4 * 256 * 2);
  _Float16* fe5  = (_Float16*)alloc((size_t)nv4 * 256 * 2);
  _Float16* M0m  = (_Float16*)alloc((size_t)nv0 * 2);
  _Float16* M1m  = (_Float16*)alloc((size_t)nv1 * 2);
  _Float16* M2m  = (_Float16*)alloc((size_t)nv2 * 2);
  _Float16* M3m  = (_Float16*)alloc((size_t)nv3 * 2);
  _Float16* M4m  = (_Float16*)alloc((size_t)nv4 * 2);

  struct LW { _Float16* wt; float* bias; };
  auto allocW = [&](int CinP, int NP) -> LW {
    LW lw;
    lw.wt   = (_Float16*)alloc((size_t)NP * 27 * CinP * 2);
    lw.bias = (float*)alloc((size_t)NP * 4);
    return lw;
  };
  LW w_e1  = allocW(32, 16);
  LW w_e2a = allocW(32, 32),  w_e2b = allocW(32, 32);
  LW w_e3a = allocW(32, 64),  w_e3b = allocW(64, 64);
  LW w_e4a = allocW(64, 128), w_e4b = allocW(128, 128);
  LW w_e5a = allocW(128, 256), w_e5b = allocW(256, 256);
  LW w_d5i = allocW(256, 128), w_d5s = allocW(128, 128);
  LW w_d4i = allocW(128, 64),  w_d4s = allocW(64, 64);
  LW w_d3i = allocW(64, 32),   w_d3s = allocW(32, 32);
  LW w_d2i = allocW(32, 16),   w_d2s = allocW(32, 16);
  LW w_d1  = allocW(32, 16);

  const size_t used = (size_t)(p - (char*)d_ws);
  hipMemsetAsync(d_ws, 0, used, stream);  // zero pads + zero page, deterministic

  // ---------------- weight/BN folding --------------------------------------
  auto prep = [&](int widx, int bnidx, LW lw, int Cin, int Cout, int CinP, int NP) {
    const long total = (long)NP * 27 * CinP;
    hipLaunchKernelGGL(prep_weights_k, dim3((unsigned)((total + 255) / 256)), dim3(256),
                       0, stream,
                       (const float*)d_in[widx], (const float*)d_in[bnidx],
                       (const float*)d_in[bnidx + 1], (const float*)d_in[bnidx + 2],
                       (const float*)d_in[bnidx + 3], lw.wt, lw.bias, Cin, Cout, CinP, NP);
  };
  prep(2, 3,   w_e1,  9,   16,  32,  16);
  prep(7, 8,   w_e2a, 16,  32,  32,  32);  prep(12, 13, w_e2b, 32,  32,  32,  32);
  prep(17, 18, w_e3a, 32,  64,  32,  64);  prep(22, 23, w_e3b, 64,  64,  64,  64);
  prep(27, 28, w_e4a, 64,  128, 64,  128); prep(32, 33, w_e4b, 128, 128, 128, 128);
  prep(37, 38, w_e5a, 128, 256, 128, 256); prep(42, 43, w_e5b, 256, 256, 256, 256);
  prep(47, 48, w_d5i, 256, 128, 256, 128); prep(52, 53, w_d5s, 128, 128, 128, 128);
  prep(59, 60, w_d4i, 128, 64,  128, 64);  prep(64, 65, w_d4s, 64,  64,  64,  64);
  prep(71, 72, w_d3i, 64,  32,  64,  32);  prep(76, 77, w_d3s, 32,  32,  32,  32);
  prep(83, 84, w_d2i, 32,  16,  32,  16);  prep(88, 89, w_d2s, 16,  16,  32,  16);
  prep(97, 98, w_d1,  16,  9,   32,  16);

  // ---------------- input + mask pyramid -----------------------------------
  hipLaunchKernelGGL(prep_input_k, dim3((unsigned)((nv0 + 255) / 256)), dim3(256), 0, stream,
                     (const float*)d_in[0], bufA, M0m, (int)nv0);
  auto down = [&](const _Float16* mi, _Float16* mo, int Di, int Hi, int Wi,
                  int Do, int Ho, int Wo) {
    const long n = (long)B * Do * Ho * Wo;
    hipLaunchKernelGGL(mask_down_k, dim3((unsigned)((n + 255) / 256)), dim3(256), 0, stream,
                       mi, mo, B, Di, Hi, Wi, Do, Ho, Wo);
  };
  down(M0m, M1m, D0, H0, W0, D1, H1, W1);
  down(M1m, M2m, D1, H1, W1, D2, H2, W2);
  down(M2m, M3m, D2, H2, W2, D3, H3, W3);
  down(M3m, M4m, D3, H3, W3, D4, H4, W4);

  // ---------------- conv launcher ------------------------------------------
  auto conv = [&](const _Float16* in, _Float16* out, LW lw, const _Float16* mask,
                  int Di, int Hi, int Wi, int CinP,
                  int Do, int Ho, int Wo, int CoutP, int NP, int mode) {
    ConvArgs a;
    a.in = in; a.out = out; a.wt = lw.wt; a.bias = lw.bias; a.mask = mask; a.zpage = zpage;
    a.B = B; a.Di = Di; a.Hi = Hi; a.Wi = Wi; a.CinP = CinP;
    a.Do = Do; a.Ho = Ho; a.Wo = Wo; a.CoutP = CoutP; a.NP = NP; a.mode = mode;
    launch_conv(a, stream);
  };

  // ---------------- encoder -------------------------------------------------
  conv(bufA, bufB, w_e1,  M0m, D0, H0, W0, 32,  D0, H0, W0, 32,  16,  0);
  conv(bufB, tmp1, w_e2a, M1m, D0, H0, W0, 32,  D1, H1, W1, 32,  32,  1);
  conv(tmp1, fe2,  w_e2b, M1m, D1, H1, W1, 32,  D1, H1, W1, 32,  32,  0);
  conv(fe2,  tmp2, w_e3a, M2m, D1, H1, W1, 32,  D2, H2, W2, 64,  64,  1);
  conv(tmp2, fe3,  w_e3b, M2m, D2, H2, W2, 64,  D2, H2, W2, 64,  64,  0);
  conv(fe3,  tmp3, w_e4a, M3m, D2, H2, W2, 64,  D3, H3, W3, 128, 128, 1);
  conv(tmp3, fe4,  w_e4b, M3m, D3, H3, W3, 128, D3, H3, W3, 128, 128, 0);
  conv(fe4,  tmp4, w_e5a, M4m, D3, H3, W3, 128, D4, H4, W4, 256, 256, 1);
  conv(tmp4, fe5,  w_e5b, M4m, D4, H4, W4, 256, D4, H4, W4, 256, 256, 0);

  // ---------------- output offsets -----------------------------------------
  float* out_preds = (float*)d_out;                      // B*D*H*W*3 points *3
  float* out_occ   = out_preds + nv0 * 9;
  float* out_p5    = out_occ + nv0;
  float* out_p4    = out_p5 + nv3;
  float* out_p3    = out_p4 + nv2;
  float* out_p2    = out_p3 + nv1;

  auto cls = [&](const _Float16* dbuf, const _Float16* mask, int widx, int bidx,
                 float* out, long nvox, int C, int CP) {
    hipLaunchKernelGGL(cls_k, dim3((unsigned)((nvox + 255) / 256)), dim3(256), 0, stream,
                       dbuf, mask, (const float*)d_in[widx], (const float*)d_in[bidx],
                       out, (int)nvox, C, CP);
  };
  auto addf = [&](_Float16* dst, const _Float16* src, long n) {
    hipLaunchKernelGGL(add_k, dim3((unsigned)((n + 255) / 256)), dim3(256), 0, stream,
                       dst, src, n);
  };

  // ---------------- decoder -------------------------------------------------
  conv(fe5,  tmp3, w_d5i, M3m, D4, H4, W4, 256, D3, H3, W3, 128, 128, 2);
  conv(tmp3, dq3,  w_d5s, M3m, D3, H3, W3, 128, D3, H3, W3, 128, 128, 0);
  cls(dq3, M3m, 57, 58, out_p5, nv3, 128, 128);
  addf(dq3, fe4, nv3 * 128);

  conv(dq3,  tmp2, w_d4i, M2m, D3, H3, W3, 128, D2, H2, W2, 64, 64, 2);
  conv(tmp2, dq2,  w_d4s, M2m, D2, H2, W2, 64,  D2, H2, W2, 64, 64, 0);
  cls(dq2, M2m, 69, 70, out_p4, nv2, 64, 64);
  addf(dq2, fe3, nv2 * 64);

  conv(dq2,  tmp1, w_d3i, M1m, D2, H2, W2, 64, D1, H1, W1, 32, 32, 2);
  conv(tmp1, dq1,  w_d3s, M1m, D1, H1, W1, 32, D1, H1, W1, 32, 32, 0);
  cls(dq1, M1m, 81, 82, out_p3, nv1, 32, 32);
  addf(dq1, fe2, nv1 * 32);

  conv(dq1,  bufA, w_d2i, M0m, D1, H1, W1, 32, D0, H0, W0, 32, 16, 2);
  conv(bufA, bufB, w_d2s, M0m, D0, H0, W0, 32, D0, H0, W0, 32, 16, 0);
  cls(bufB, M0m, 93, 94, out_p2, nv0, 16, 32);

  // ---------------- heads ---------------------------------------------------
  hipLaunchKernelGGL(occ_k, dim3((unsigned)((nv0 + 255) / 256)), dim3(256), 0, stream,
                     bufB, (const float*)d_in[95], (const float*)d_in[96], out_occ,
                     B, D0, H0, W0);
  conv(bufB, bufA, w_d1, M0m, D0, H0, W0, 32, D0, H0, W0, 32, 16, 0);
  hipLaunchKernelGGL(preds_k, dim3((unsigned)((nv0 + 255) / 256)), dim3(256), 0, stream,
                     bufA, out_preds, B, D0, H0, W0);
}
